// BertCRFModel_17772574671431
// MI455X (gfx1250) — matrix operand order repaired and checked by reading, not compiled, not run on hardware
//
#include <hip/hip_runtime.h>
#include <hip/hip_bf16.h>
#include <math.h>

#define BB   16
#define SS   256
#define HH   768
#define LLY  12
#define NHD  12
#define DHD  64
#define FFD  3072
#define TTG  9

typedef __attribute__((ext_vector_type(16))) __bf16 v16bf;
typedef __attribute__((ext_vector_type(8)))  __bf16 v8bf;
typedef __attribute__((ext_vector_type(8)))  float  v8f;

__device__ inline __bf16 f2bf(float f) {
    union { float f; unsigned u; } a; a.f = f;
    unsigned r = (a.u + 0x7FFFu + ((a.u >> 16) & 1u)) >> 16;   // round-to-nearest-even
    union { unsigned short s; __bf16 b; } o; o.s = (unsigned short)r;
    return o.b;
}

__device__ inline float wave_sum(float v) {
    #pragma unroll
    for (int m = 16; m > 0; m >>= 1) v += __shfl_xor(v, m, 32);
    return v;
}
__device__ inline float wave_max(float v) {
    #pragma unroll
    for (int m = 16; m > 0; m >>= 1) v = fmaxf(v, __shfl_xor(v, m, 32));
    return v;
}

// ---------------------------------------------------------------------------
// Weight convert + transpose: in fp32 [z][K,N] -> out bf16 [z][N,K]
// 32x32 LDS tile for coalesced read AND write.
// ---------------------------------------------------------------------------
__global__ void wtrans_kernel(const float* __restrict__ in, __bf16* __restrict__ out,
                              int K, int N) {
    __shared__ __bf16 tile[32][33];
    size_t zb = (size_t)blockIdx.z * K * N;
    int nt = blockIdx.x * 32, kt = blockIdx.y * 32;
    #pragma unroll
    for (int j = 0; j < 4; ++j) {
        int kk = kt + threadIdx.y + 8 * j;
        tile[threadIdx.y + 8 * j][threadIdx.x] = f2bf(in[zb + (size_t)kk * N + nt + threadIdx.x]);
    }
    __syncthreads();
    #pragma unroll
    for (int j = 0; j < 4; ++j) {
        int nn = nt + threadIdx.y + 8 * j;
        out[zb + (size_t)nn * K + kt + threadIdx.x] = tile[threadIdx.x][threadIdx.y + 8 * j];
    }
}

// ---------------------------------------------------------------------------
// Batched WMMA GEMM, all-NT, pure bf16 operand path:
//   C = scale * (A @ B^T) + bias ; A bf16 [M,K] lda ; B bf16 [N,K] ldb
//   cmode: 0 -> fp32 C[m*ldc+n]; 1 -> bf16 C[m*ldc+n]; 2 -> bf16 C[n*ldc+m]
// Wave computes 32(M) x 32(N) = four 16x16 tiles; B fragments reused across
// two A fragments (2 loads per WMMA). All fragment loads contiguous b128.
// ---------------------------------------------------------------------------
__global__ void wmma_gemm_kernel(const __bf16* __restrict__ A, const __bf16* __restrict__ B,
                                 const float* __restrict__ bias, void* __restrict__ Cout,
                                 int M, int N, int K, int lda, int ldb, int ldc,
                                 int cmode, float scale, int nh,
                                 int sAb, int sAh, int sBb, int sBh, int sCb, int sCh) {
    int z = blockIdx.z;
    int bb = z / nh, hh = z % nh;
    A += (size_t)bb * sAb + (size_t)hh * sAh;
    B += (size_t)bb * sBb + (size_t)hh * sBh;
    size_t coff = (size_t)bb * sCb + (size_t)hh * sCh;

    int lane  = threadIdx.x;
    int tileM = blockIdx.y * 8 + threadIdx.y;      // 32-row strip index
    int l15   = lane & 15;
    int n0    = blockIdx.x * 32 + l15, n1 = n0 + 16;
    int row0  = tileM * 32 + l15, row1 = row0 + 16;
    int hi8   = (lane >> 4) << 3;    // A frag: K groups interleaved by lane half
    int khl   = (lane >> 4) << 3;    // B frag: K groups interleaved by lane half (same as A)

    const __bf16* Ar0 = A + (size_t)row0 * lda + hi8;
    const __bf16* Ar1 = A + (size_t)row1 * lda + hi8;
    const __bf16* Br0 = B + (size_t)n0  * ldb + khl;
    const __bf16* Br1 = B + (size_t)n1  * ldb + khl;

    v8f acc00 = {}; v8f acc01 = {}; v8f acc10 = {}; v8f acc11 = {};
    #pragma unroll 2
    for (int k0 = 0; k0 < K; k0 += 32) {
        v8bf t0, t1;
        t0 = *(const v8bf*)(Ar0 + k0);               // K = k0+hi8 .. +7
        t1 = *(const v8bf*)(Ar0 + k0 + 16);          // K = k0+hi8+16 .. +23
        v16bf a0 = __builtin_shufflevector(t0, t1, 0,1,2,3,4,5,6,7,8,9,10,11,12,13,14,15);
        t0 = *(const v8bf*)(Ar1 + k0);
        t1 = *(const v8bf*)(Ar1 + k0 + 16);
        v16bf a1 = __builtin_shufflevector(t0, t1, 0,1,2,3,4,5,6,7,8,9,10,11,12,13,14,15);
        t0 = *(const v8bf*)(Br0 + k0);
        t1 = *(const v8bf*)(Br0 + k0 + 16);
        v16bf b0 = __builtin_shufflevector(t0, t1, 0,1,2,3,4,5,6,7,8,9,10,11,12,13,14,15);
        t0 = *(const v8bf*)(Br1 + k0);
        t1 = *(const v8bf*)(Br1 + k0 + 16);
        v16bf b1 = __builtin_shufflevector(t0, t1, 0,1,2,3,4,5,6,7,8,9,10,11,12,13,14,15);
        acc00 = __builtin_amdgcn_wmma_f32_16x16x32_bf16(false, a0, false, b0, (short)0, acc00, false, false);
        acc01 = __builtin_amdgcn_wmma_f32_16x16x32_bf16(false, a0, false, b1, (short)0, acc01, false, false);
        acc10 = __builtin_amdgcn_wmma_f32_16x16x32_bf16(false, a1, false, b0, (short)0, acc10, false, false);
        acc11 = __builtin_amdgcn_wmma_f32_16x16x32_bf16(false, a1, false, b1, (short)0, acc11, false, false);
    }

    float add0 = bias ? bias[n0] : 0.f;
    float add1 = bias ? bias[n1] : 0.f;
    if (cmode == 0) {
        float* C = (float*)Cout + coff;
        #pragma unroll
        for (int r = 0; r < 8; ++r) {
            int cm0 = tileM * 32 + r + hi8, cm1 = cm0 + 16;
            C[(size_t)cm0 * ldc + n0] = acc00[r] * scale + add0;
            C[(size_t)cm0 * ldc + n1] = acc01[r] * scale + add1;
            C[(size_t)cm1 * ldc + n0] = acc10[r] * scale + add0;
            C[(size_t)cm1 * ldc + n1] = acc11[r] * scale + add1;
        }
    } else if (cmode == 1) {
        __bf16* C = (__bf16*)Cout + coff;
        #pragma unroll
        for (int r = 0; r < 8; ++r) {
            int cm0 = tileM * 32 + r + hi8, cm1 = cm0 + 16;
            C[(size_t)cm0 * ldc + n0] = f2bf(acc00[r] * scale + add0);
            C[(size_t)cm0 * ldc + n1] = f2bf(acc01[r] * scale + add1);
            C[(size_t)cm1 * ldc + n0] = f2bf(acc10[r] * scale + add0);
            C[(size_t)cm1 * ldc + n1] = f2bf(acc11[r] * scale + add1);
        }
    } else {                                          // transposed bf16 store (for v^T)
        __bf16* C = (__bf16*)Cout + coff;
        #pragma unroll
        for (int r = 0; r < 8; ++r) {
            int cm0 = tileM * 32 + r + hi8, cm1 = cm0 + 16;
            C[(size_t)n0 * ldc + cm0] = f2bf(acc00[r] * scale + add0);
            C[(size_t)n1 * ldc + cm0] = f2bf(acc01[r] * scale + add1);
            C[(size_t)n0 * ldc + cm1] = f2bf(acc10[r] * scale + add0);
            C[(size_t)n1 * ldc + cm1] = f2bf(acc11[r] * scale + add1);
        }
    }
}

// ---------------------------------------------------------------------------
// Embedding gather + LayerNorm; writes fp32 x and bf16 mirror
// ---------------------------------------------------------------------------
__global__ void embed_ln_kernel(const int* __restrict__ ids, const float* __restrict__ ew,
                                const float* __restrict__ ep, const float* __restrict__ et,
                                const float* __restrict__ g, const float* __restrict__ be,
                                float* __restrict__ x, __bf16* __restrict__ xb) {
    int row  = blockIdx.x * 8 + threadIdx.y;
    int lane = threadIdx.x;
    int spos = row & (SS - 1);
    const float* wr = ew + (size_t)ids[row] * HH;
    const float* pr = ep + (size_t)spos * HH;
    float v[24]; float s = 0.f;
    #pragma unroll
    for (int j = 0; j < 24; ++j) { int c = lane + 32 * j; float t = wr[c] + pr[c] + et[c]; v[j] = t; s += t; }
    s = wave_sum(s); float mean = s * (1.f / HH);
    float s2 = 0.f;
    #pragma unroll
    for (int j = 0; j < 24; ++j) { float d = v[j] - mean; s2 += d * d; }
    s2 = wave_sum(s2);
    float inv = rsqrtf(s2 * (1.f / HH) + 1e-12f);
    #pragma unroll
    for (int j = 0; j < 24; ++j) {
        int c = lane + 32 * j;
        float o = (v[j] - mean) * inv * g[c] + be[c];
        x[(size_t)row * HH + c]  = o;
        xb[(size_t)row * HH + c] = f2bf(o);
    }
}

// x = LN(x + y); writes fp32 x and bf16 mirror
__global__ void add_ln_kernel(float* __restrict__ x, const float* __restrict__ y,
                              const float* __restrict__ g, const float* __restrict__ be,
                              __bf16* __restrict__ xb) {
    int row  = blockIdx.x * 8 + threadIdx.y;
    int lane = threadIdx.x;
    float v[24]; float s = 0.f;
    #pragma unroll
    for (int j = 0; j < 24; ++j) {
        int c = lane + 32 * j; size_t o = (size_t)row * HH + c;
        float t = x[o] + y[o]; v[j] = t; s += t;
    }
    s = wave_sum(s); float mean = s * (1.f / HH);
    float s2 = 0.f;
    #pragma unroll
    for (int j = 0; j < 24; ++j) { float d = v[j] - mean; s2 += d * d; }
    s2 = wave_sum(s2);
    float inv = rsqrtf(s2 * (1.f / HH) + 1e-12f);
    #pragma unroll
    for (int j = 0; j < 24; ++j) {
        int c = lane + 32 * j;
        float o = (v[j] - mean) * inv * g[c] + be[c];
        x[(size_t)row * HH + c]  = o;
        xb[(size_t)row * HH + c] = f2bf(o);
    }
}

// masked softmax: fp32 scores in, bf16 probs out (one wave per (b,h,q) row)
__global__ void softmax_kernel(const float* __restrict__ sc, __bf16* __restrict__ pb,
                               const int* __restrict__ am) {
    int row  = blockIdx.x * 8 + threadIdx.y;
    int lane = threadIdx.x;
    int b    = row / (NHD * SS);
    const float* r = sc + (size_t)row * SS;
    __bf16* o = pb + (size_t)row * SS;
    const int* m = am + b * SS;
    float v[8]; float mx = -3.0e38f;
    #pragma unroll
    for (int j = 0; j < 8; ++j) {
        int c = lane + 32 * j;
        float t = r[c] + (m[c] > 0 ? 0.f : -1e4f);
        v[j] = t; mx = fmaxf(mx, t);
    }
    mx = wave_max(mx);
    float s = 0.f;
    #pragma unroll
    for (int j = 0; j < 8; ++j) { v[j] = expf(v[j] - mx); s += v[j]; }
    s = wave_sum(s);
    float inv = 1.f / s;
    #pragma unroll
    for (int j = 0; j < 8; ++j) o[lane + 32 * j] = f2bf(v[j] * inv);
}

// GELU: fp32 in, bf16 out
__global__ void gelu_kernel(const float* __restrict__ h, __bf16* __restrict__ hb, long long n) {
    long long i = (long long)blockIdx.x * blockDim.x + threadIdx.x;
    if (i < n) { float x = h[i]; hb[i] = f2bf(0.5f * x * (1.f + erff(x * 0.70710678118654752f))); }
}

// logits = hid @ W_tag + b_tag   (T=9, tiny)
__global__ void tag_logits_kernel(const float* __restrict__ hid, const float* __restrict__ Wt,
                                  const float* __restrict__ bt, float* __restrict__ out, int Mrows) {
    int idx = blockIdx.x * blockDim.x + threadIdx.x;
    int m = idx / TTG, t = idx % TTG;
    if (m >= Mrows) return;
    const float* hr = hid + (size_t)m * HH;
    float s = bt[t];
    for (int k = 0; k < HH; ++k) s += hr[k] * Wt[k * TTG + t];
    out[(size_t)m * TTG + t] = s;
}

// CRF log-likelihood, one block (1 wave) per batch element
__global__ void crf_loss_kernel(const float* __restrict__ em, const int* __restrict__ labels,
                                const int* __restrict__ amask, const float* __restrict__ start,
                                const float* __restrict__ endv, const float* __restrict__ tr,
                                float* __restrict__ llh) {
    int b = blockIdx.x, j = threadIdx.x;
    __shared__ float alpha[TTG], nalpha[TTG], trs[TTG * TTG];
    for (int i = j; i < TTG * TTG; i += 32) trs[i] = tr[i];
    const float* E  = em + (size_t)b * SS * TTG;
    const int*   am = amask + b * SS;
    if (j < TTG) alpha[j] = start[j] + E[j];
    __syncthreads();
    for (int t = 1; t < SS; ++t) {
        bool m = am[t] > 0;
        if (j < TTG) {
            float mx = -3.0e38f;
            #pragma unroll
            for (int i = 0; i < TTG; ++i) mx = fmaxf(mx, alpha[i] + trs[i * TTG + j]);
            float s = 0.f;
            #pragma unroll
            for (int i = 0; i < TTG; ++i) s += expf(alpha[i] + trs[i * TTG + j] - mx);
            float nv = mx + logf(s) + E[t * TTG + j];
            nalpha[j] = m ? nv : alpha[j];
        }
        __syncthreads();
        if (j < TTG) alpha[j] = nalpha[j];
        __syncthreads();
    }
    if (j == 0) {
        float mx = -3.0e38f;
        for (int i = 0; i < TTG; ++i) mx = fmaxf(mx, alpha[i] + endv[i]);
        float s = 0.f;
        for (int i = 0; i < TTG; ++i) s += expf(alpha[i] + endv[i] - mx);
        float logZ = mx + logf(s);
        const int* lb = labels + b * SS;
        float sc = start[lb[0]] + E[lb[0]];
        int cnt = 0;
        for (int t = 0; t < SS; ++t) cnt += (am[t] > 0);
        int prev = lb[0];
        for (int t = 1; t < SS; ++t) {
            float mf = am[t] > 0 ? 1.f : 0.f;
            sc += (trs[prev * TTG + lb[t]] + E[t * TTG + lb[t]]) * mf;
            prev = lb[t];
        }
        sc += endv[lb[cnt - 1]];
        llh[b] = sc - logZ;
    }
}

__global__ void loss_finalize_kernel(const float* __restrict__ llh, float* __restrict__ out) {
    float s = 0.f;
    for (int b = 0; b < BB; ++b) s += llh[b];
    out[0] = -s;
}

// Viterbi decode, one block (1 wave) per batch element; preds written as float
__global__ void viterbi_kernel(const float* __restrict__ em, const int* __restrict__ amask,
                               const float* __restrict__ start, const float* __restrict__ endv,
                               const float* __restrict__ tr, int* __restrict__ ptrs,
                               float* __restrict__ preds) {
    int b = blockIdx.x, j = threadIdx.x;
    __shared__ float alpha[TTG], nalpha[TTG], trs[TTG * TTG];
    for (int i = j; i < TTG * TTG; i += 32) trs[i] = tr[i];
    const float* E  = em + (size_t)b * SS * TTG;
    const int*   am = amask + b * SS;
    int* P = ptrs + (size_t)b * (SS - 1) * TTG;
    if (j < TTG) alpha[j] = start[j] + E[j];
    __syncthreads();
    for (int t = 1; t < SS; ++t) {
        bool m = am[t] > 0;
        if (j < TTG) {
            float best = -3.0e38f; int bi = 0;
            #pragma unroll
            for (int i = 0; i < TTG; ++i) {
                float s = alpha[i] + trs[i * TTG + j];
                if (s > best) { best = s; bi = i; }
            }
            nalpha[j] = m ? best + E[t * TTG + j] : alpha[j];
            P[(t - 1) * TTG + j] = m ? bi : j;
        }
        __syncthreads();
        if (j < TTG) alpha[j] = nalpha[j];
        __syncthreads();
    }
    if (j == 0) {
        float best = -3.0e38f; int tag = 0;
        for (int i = 0; i < TTG; ++i) {
            float s = alpha[i] + endv[i];
            if (s > best) { best = s; tag = i; }
        }
        float* o = preds + (size_t)b * SS;
        o[SS - 1] = am[SS - 1] > 0 ? (float)tag : 0.f;
        for (int t = SS - 1; t >= 1; --t) {
            tag = P[(t - 1) * TTG + tag];
            o[t - 1] = am[t - 1] > 0 ? (float)tag : 0.f;
        }
    }
}

// ---------------------------------------------------------------------------
static void launch_gemm(const __bf16* A, const __bf16* B, const float* bias, void* C,
                        int M, int N, int K, int lda, int ldb, int ldc, int cmode, float scale,
                        int batch, int nh, int sAb, int sAh, int sBb, int sBh, int sCb, int sCh,
                        hipStream_t st) {
    dim3 grid(N / 32, M / 256, batch);   // wave = 32M x 32N, 8 waves/block
    dim3 block(32, 8, 1);
    wmma_gemm_kernel<<<grid, block, 0, st>>>(A, B, bias, C, M, N, K, lda, ldb, ldc,
                                             cmode, scale, nh, sAb, sAh, sBb, sBh, sCb, sCh);
}

extern "C" void kernel_launch(void* const* d_in, const int* in_sizes, int n_in,
                              void* d_out, int out_size, void* d_ws, size_t ws_size,
                              hipStream_t stream) {
    (void)in_sizes; (void)n_in; (void)out_size; (void)ws_size;
    const int*   ids      = (const int*)d_in[0];
    const int*   amask    = (const int*)d_in[1];
    const int*   labels   = (const int*)d_in[2];
    const float* emb_word = (const float*)d_in[3];
    const float* emb_pos  = (const float*)d_in[4];
    const float* emb_type = (const float*)d_in[5];
    const float* lng      = (const float*)d_in[6];
    const float* lnb      = (const float*)d_in[7];
    const float* Wq = (const float*)d_in[8];  const float* bq = (const float*)d_in[9];
    const float* Wk = (const float*)d_in[10]; const float* bk = (const float*)d_in[11];
    const float* Wv = (const float*)d_in[12]; const float* bv = (const float*)d_in[13];
    const float* Wo = (const float*)d_in[14]; const float* bo = (const float*)d_in[15];
    const float* ln1g = (const float*)d_in[16]; const float* ln1b = (const float*)d_in[17];
    const float* W1 = (const float*)d_in[18]; const float* b1 = (const float*)d_in[19];
    const float* W2 = (const float*)d_in[20]; const float* b2 = (const float*)d_in[21];
    const float* ln2g = (const float*)d_in[22]; const float* ln2b = (const float*)d_in[23];
    const float* Wt = (const float*)d_in[24]; const float* bt = (const float*)d_in[25];
    const float* cstart = (const float*)d_in[26];
    const float* cend   = (const float*)d_in[27];
    const float* ctr    = (const float*)d_in[28];

    float* out = (float*)d_out;
    float* w   = (float*)d_ws;

    const size_t NX = (size_t)BB * SS * HH;        // 3,145,728
    const size_t NS = (size_t)BB * NHD * SS * SS;  // 12,582,912
    const size_t NF = (size_t)BB * SS * FFD;       // 12,582,912
    const size_t WHH = (size_t)LLY * HH * HH;      // 7,077,888
    const size_t WHF = (size_t)LLY * HH * FFD;     // 28,311,552

    // fp32 region
    float* x    = w;                               // [M,H]
    float* scf  = x   + NX;                        // scores [B,NH,S,S]
    float* y    = scf + NS;                        // gemm fp32 out
    float* h    = y   + NX;                        // FFN intermediate
    float* lgts = h   + NF;                        // [M,T]
    float* llh  = lgts + (size_t)BB * SS * TTG;
    int*   ptrs = (int*)(llh + BB);                // [B,S-1,T]
    // bf16 region (16B aligned: all prior counts are multiples of 4 floats)
    __bf16* xb   = (__bf16*)(ptrs + (size_t)BB * (SS - 1) * TTG);
    __bf16* qb   = xb   + NX;
    __bf16* kb   = qb   + NX;
    __bf16* vtb  = kb   + NX;                      // v^T: [H, M] = [768, 4096]
    __bf16* ctxb = vtb  + NX;
    __bf16* pb   = ctxb + NX;                      // probs bf16 [B,NH,S,S]
    __bf16* hb   = pb   + NS;                      // gelu(h) bf16
    __bf16* Wqb  = hb   + NF;                      // [L][H(out) , H(in)] (transposed)
    __bf16* Wkb  = Wqb + WHH;
    __bf16* Wvb  = Wkb + WHH;
    __bf16* Wob  = Wvb + WHH;
    __bf16* W1b  = Wob + WHH;                      // [L][FF, H]
    __bf16* W2b  = W1b + WHF;                      // [L][H, FF]

    const int M = BB * SS;                         // 4096
    dim3 rows8(M / 8), smrows((BB * NHD * SS) / 8), wav(32, 8, 1);
    dim3 tb(32, 8, 1);

    // one-shot weight convert+transpose (bandwidth-trivial vs GEMM compute)
    wtrans_kernel<<<dim3(HH / 32,  HH / 32,  LLY), tb, 0, stream>>>(Wq, Wqb, HH,  HH);
    wtrans_kernel<<<dim3(HH / 32,  HH / 32,  LLY), tb, 0, stream>>>(Wk, Wkb, HH,  HH);
    wtrans_kernel<<<dim3(HH / 32,  HH / 32,  LLY), tb, 0, stream>>>(Wv, Wvb, HH,  HH);
    wtrans_kernel<<<dim3(HH / 32,  HH / 32,  LLY), tb, 0, stream>>>(Wo, Wob, HH,  HH);
    wtrans_kernel<<<dim3(FFD / 32, HH / 32,  LLY), tb, 0, stream>>>(W1, W1b, HH,  FFD);
    wtrans_kernel<<<dim3(HH / 32,  FFD / 32, LLY), tb, 0, stream>>>(W2, W2b, FFD, HH);

    embed_ln_kernel<<<rows8, wav, 0, stream>>>(ids, emb_word, emb_pos, emb_type, lng, lnb, x, xb);

    const int sQKb = SS * HH, sQKh = DHD;          // bf16 [B,S,H] batched by (b, head)
    const int sSCb = NHD * SS * SS, sSCh = SS * SS;
    const int sVTb = SS, sVTh = DHD * M;           // v^T [H, M]

    for (int l = 0; l < LLY; ++l) {
        const __bf16* Wql = Wqb + (size_t)l * HH * HH;  const float* bql = bq + (size_t)l * HH;
        const __bf16* Wkl = Wkb + (size_t)l * HH * HH;  const float* bkl = bk + (size_t)l * HH;
        const __bf16* Wvl = Wvb + (size_t)l * HH * HH;  const float* bvl = bv + (size_t)l * HH;
        const __bf16* Wol = Wob + (size_t)l * HH * HH;  const float* bol = bo + (size_t)l * HH;
        const __bf16* W1l = W1b + (size_t)l * HH * FFD; const float* b1l = b1 + (size_t)l * FFD;
        const __bf16* W2l = W2b + (size_t)l * FFD * HH; const float* b2l = b2 + (size_t)l * HH;

        // QKV projections -> bf16 (v stored transposed [H, M])
        launch_gemm(xb, Wql, bql, qb,  M, HH, HH, HH, HH, HH, 1, 1.f, 1, 1, 0,0,0,0,0,0, stream);
        launch_gemm(xb, Wkl, bkl, kb,  M, HH, HH, HH, HH, HH, 1, 1.f, 1, 1, 0,0,0,0,0,0, stream);
        launch_gemm(xb, Wvl, bvl, vtb, M, HH, HH, HH, HH, M,  2, 1.f, 1, 1, 0,0,0,0,0,0, stream);
        // scores = q @ k^T * 1/sqrt(DH)  (fp32 out), batched over (b,h)
        launch_gemm(qb, kb, nullptr, scf, SS, SS, DHD, HH, HH, SS, 0, 0.125f,
                    BB * NHD, NHD, sQKb, sQKh, sQKb, sQKh, sSCb, sSCh, stream);
        softmax_kernel<<<smrows, wav, 0, stream>>>(scf, pb, amask);
        // ctx = attn @ v  ==  attn @ (v^T)^T  -> bf16 [M,H]
        launch_gemm(pb, vtb, nullptr, ctxb, SS, DHD, SS, SS, M, HH, 1, 1.f,
                    BB * NHD, NHD, sSCb, sSCh, sVTb, sVTh, sQKb, sQKh, stream);
        // output projection (fp32) + residual LN
        launch_gemm(ctxb, Wol, bol, y, M, HH, HH, HH, HH, HH, 0, 1.f, 1, 1, 0,0,0,0,0,0, stream);
        add_ln_kernel<<<rows8, wav, 0, stream>>>(x, y, ln1g + (size_t)l * HH, ln1b + (size_t)l * HH, xb);
        // FFN
        launch_gemm(xb, W1l, b1l, h, M, FFD, HH, HH, HH, FFD, 0, 1.f, 1, 1, 0,0,0,0,0,0, stream);
        gelu_kernel<<<(unsigned)((NF + 255) / 256), 256, 0, stream>>>(h, hb, (long long)NF);
        launch_gemm(hb, W2l, b2l, y, M, HH, FFD, FFD, FFD, HH, 0, 1.f, 1, 1, 0,0,0,0,0,0, stream);
        add_ln_kernel<<<rows8, wav, 0, stream>>>(x, y, ln2g + (size_t)l * HH, ln2b + (size_t)l * HH, xb);
    }

    tag_logits_kernel<<<(M * TTG + 255) / 256, 256, 0, stream>>>(x, Wt, bt, lgts, M);
    crf_loss_kernel<<<BB, 32, 0, stream>>>(lgts, labels, amask, cstart, cend, ctr, llh);
    loss_finalize_kernel<<<1, 1, 0, stream>>>(llh, out);
    viterbi_kernel<<<BB, 32, 0, stream>>>(lgts, amask, cstart, cend, ctr, ptrs, out + 1);
}
